// SegmentCausalCrossAttention_36043365548100
// MI455X (gfx1250) — matrix operand order repaired and checked by reading, not compiled, not run on hardware
//
#include <hip/hip_runtime.h>
#include <hip/hip_bf16.h>
#include <math.h>

// ---------------------------------------------------------------------------
// SegmentCausalCrossAttention for MI455X (gfx1250, wave32, WMMA, async-LDS)
//   B=2, Lq=4096, Lkv=512, H=16, DH=64, LOOKBACK=7 (Kw=8), SMAX=8192
//
//   prep:     q, kv_src -> bf16;  Wq, Wkv, Wo -> bf16 transposed [n][k]
//   gemm:     double-buffered LDS tiles staged with
//             global_load_async_to_lds_b128 (ASYNCcnt), fragments via
//             ds_load_b128, compute via v_wmma_f32_16x16x32_bf16
//   attn:     wave32 per (b,q,h) gather + RoPE + 8-slot softmax, bf16 out
// ---------------------------------------------------------------------------

typedef __attribute__((ext_vector_type(4)))  __bf16 bf16x4;
typedef __attribute__((ext_vector_type(8)))  __bf16 bf16x8;
typedef __attribute__((ext_vector_type(16))) __bf16 bf16x16;
typedef __attribute__((ext_vector_type(8)))  float  f32x8;

#define CC_B     2
#define CC_LQ    4096
#define CC_LKV   512
#define CC_H     16
#define CC_DH    64
#define CC_KW    8
#define CC_SMAX  8192

static __device__ __forceinline__ bf16x16 combine16(bf16x8 lo, bf16x8 hi) {
  return __builtin_shufflevector(lo, hi, 0, 1, 2, 3, 4, 5, 6, 7,
                                         8, 9, 10, 11, 12, 13, 14, 15);
}

// 16-byte async copy global -> LDS (VGLOBAL encoding, GV mode, ASYNCcnt).
static __device__ __forceinline__ void async_copy_b128(unsigned lds_off,
                                                       const void* gaddr) {
  asm volatile("global_load_async_to_lds_b128 %0, %1, off"
               :: "v"(lds_off), "v"(gaddr) : "memory");
}

// ---------------------------------------------------------------------------
// Elementwise f32 -> bf16 (vectorized by 4)
// ---------------------------------------------------------------------------
__global__ __launch_bounds__(256)
void convert_f32_to_bf16(const float* __restrict__ src, __bf16* __restrict__ dst,
                         int n4) {
  const int i = blockIdx.x * blockDim.x + threadIdx.x;
  if (i < n4) {
    const float4 v = ((const float4*)src)[i];
    bf16x4 pk = { (__bf16)v.x, (__bf16)v.y, (__bf16)v.z, (__bf16)v.w };
    ((bf16x4*)dst)[i] = pk;
  }
}

// ---------------------------------------------------------------------------
// W[K][N] f32 -> Wt[N][K] bf16, LDS-tiled 32x32 transpose.
// grid = (N/32, K/32), block = 256.
// ---------------------------------------------------------------------------
__global__ __launch_bounds__(256)
void transpose_f32_to_bf16(const float* __restrict__ W, __bf16* __restrict__ Wt,
                           int K, int N) {
  __shared__ float lt[32][33];
  const int k0 = blockIdx.y * 32;
  const int n0 = blockIdx.x * 32;
  const int r  = threadIdx.x >> 3;        // 0..31
  const int c4 = (threadIdx.x & 7) * 4;   // 0,4,..,28

  const float4 v = *(const float4*)(W + (size_t)(k0 + r) * N + n0 + c4);
  lt[r][c4 + 0] = v.x; lt[r][c4 + 1] = v.y;
  lt[r][c4 + 2] = v.z; lt[r][c4 + 3] = v.w;
  __syncthreads();

  bf16x4 pk = { (__bf16)lt[c4 + 0][r], (__bf16)lt[c4 + 1][r],
                (__bf16)lt[c4 + 2][r], (__bf16)lt[c4 + 3][r] };
  *(bf16x4*)&Wt[(size_t)(n0 + r) * K + k0 + c4] = pk;
}

// ---------------------------------------------------------------------------
// GEMM: Y[M,N] f32 = Xbf[M,K] (bf16 row-major) @ Wt[N,K] (bf16, K-contiguous).
// Block: 256 threads = 8 waves; block tile 128M x 128N; wave tile 32M x 64N.
// Double-buffered LDS tiles staged with async b128 copies.
// Requires M%128==0, N%128==0, K%32==0.
// ---------------------------------------------------------------------------
__global__ __launch_bounds__(256)
void gemm_bf16_wmma_async(const __bf16* __restrict__ Xbf,
                          const __bf16* __restrict__ Wt,
                          float* __restrict__ Y, int M, int N, int K) {
  __shared__ __bf16 lA[2][128 * 32];   // [m][k] tile, 8 KB each
  __shared__ __bf16 lB[2][128 * 32];   // [n][k] tile, 8 KB each

  const int tid  = threadIdx.x;
  const int lane = tid & 31;
  const int w    = tid >> 5;
  const int l15  = lane & 15;
  const int hi   = lane >> 4;
  const int wm   = (w & 3) * 32;
  const int wn   = (w >> 2) * 64;
  const int bM   = blockIdx.y * 128;
  const int bN   = blockIdx.x * 128;

  // Each thread stages 2 x 16B chunks of A and 2 x 16B of B per K-step.
  // Chunk c in 0..511: row = c>>2 (64B rows), byte offset = (c&3)*16.
  const int c0   = tid * 2;
  const int rowA = c0 >> 2;
  const int boff = (c0 & 3) * 16;      // 0 or 32 (chunks c0,c0+1 -> boff,boff+16)

  auto stage = [&](int buf, int k0) {
    const unsigned aB = (unsigned)(uintptr_t)&lA[buf][0];
    const unsigned bB = (unsigned)(uintptr_t)&lB[buf][0];
    const __bf16* ga = Xbf + (size_t)(bM + rowA) * K + k0 + (boff >> 1);
    const __bf16* gb = Wt  + (size_t)(bN + rowA) * K + k0 + (boff >> 1);
    async_copy_b128(aB + rowA * 64 + boff,      ga);
    async_copy_b128(aB + rowA * 64 + boff + 16, ga + 8);
    async_copy_b128(bB + rowA * 64 + boff,      gb);
    async_copy_b128(bB + rowA * 64 + boff + 16, gb + 8);
  };

  f32x8 acc[2][4] = {};

  stage(0, 0);                      // prologue
  int buf = 0;
  for (int k0 = 0; k0 < K; k0 += 32) {
    const bool more = (k0 + 32) < K;
    if (more) stage(buf ^ 1, k0 + 32);

    // ASYNCcnt decrements in order: <=4 outstanding => current buffer landed.
    if (more) asm volatile("s_wait_asynccnt 0x4" ::: "memory");
    else      asm volatile("s_wait_asynccnt 0x0" ::: "memory");
    __syncthreads();

    // ---- A fragments (16x32 bf16, ISA 7.12.2): half0 K0..7/16..23, half1 +8
    bf16x16 afrag[2];
#pragma unroll
    for (int mt = 0; mt < 2; ++mt) {
      const __bf16* p = &lA[buf][(wm + mt * 16 + l15) * 32 + hi * 8];
      afrag[mt] = combine16(*(const bf16x8*)p, *(const bf16x8*)(p + 16));
    }

    // ---- B fragments (32x16 bf16): half0 K0..7/16..23, half1 +8, col=lane&15
#pragma unroll
    for (int t = 0; t < 4; ++t) {
      const __bf16* p = &lB[buf][(wn + t * 16 + l15) * 32 + hi * 8];
      bf16x16 bfrag = combine16(*(const bf16x8*)p, *(const bf16x8*)(p + 16));
#pragma unroll
      for (int mt = 0; mt < 2; ++mt) {
        acc[mt][t] = __builtin_amdgcn_wmma_f32_16x16x32_bf16(
            false, afrag[mt], false, bfrag, (short)0, acc[mt][t], false, false);
      }
    }
    __syncthreads();   // all waves done reading buf before it is restaged
    buf ^= 1;
  }

  // ---- store: C/D layout: reg r -> M = r + 8*hi, N = lane&15 ----
#pragma unroll
  for (int mt = 0; mt < 2; ++mt) {
#pragma unroll
    for (int t = 0; t < 4; ++t) {
      const int n = bN + wn + t * 16 + l15;
      const int m = bM + wm + mt * 16 + hi * 8;
#pragma unroll
      for (int r = 0; r < 8; ++r) {
        Y[(size_t)(m + r) * N + n] = acc[mt][t][r];
      }
    }
  }
}

// ---------------------------------------------------------------------------
// Windowed-segment attention, one wave32 per (b, q, h).
// Lane l owns head dims (l, l+32) -> RoPE half-pairing is per-lane.
// Reads f32 qh and kv; writes bf16 attention output (feeds final WMMA GEMM).
// kv rows: [0..1023] = K projection, [1024..2047] = V projection.
// ---------------------------------------------------------------------------
__global__ __launch_bounds__(256)
void attn_window_kernel(const float* __restrict__ qh_in,
                        __bf16* __restrict__ attn_out,
                        const float* __restrict__ kv,
                        const int* __restrict__ seg_id,
                        const int* __restrict__ q_pos_ids,
                        const int* __restrict__ kv_pos_ids) {
  const int tid  = blockIdx.x * blockDim.x + threadIdx.x;
  const int wave = tid >> 5;
  const int lane = tid & 31;
  if (wave >= CC_B * CC_LQ * CC_H) return;

  const int h  = wave % CC_H;
  const int t2 = wave / CC_H;
  const int qi = t2 % CC_LQ;
  const int b  = t2 / CC_LQ;

  // inv_freq for this lane's rotary pair: 10000^(-lane/32)
  const float inv_freq = __expf(-(float)lane * 0.28782313662425572f); // ln(1e4)/32

  int qp = q_pos_ids[b * CC_LQ + qi];
  qp = qp < 0 ? 0 : (qp > CC_SMAX - 1 ? CC_SMAX - 1 : qp);
  const float aq = (float)qp * inv_freq;
  const float cq = __cosf(aq), sq = __sinf(aq);

  const size_t qoff = ((size_t)(b * CC_LQ + qi)) * (CC_H * CC_DH) + h * CC_DH;
  const float q1 = qh_in[qoff + lane];
  const float q2 = qh_in[qoff + 32 + lane];
  const float qr1 = q1 * cq - q2 * sq;
  const float qr2 = q2 * cq + q1 * sq;

  const int sid = seg_id[b * CC_LQ + qi];

  float sc[CC_KW], v1r[CC_KW], v2r[CC_KW];
  unsigned negmask = 0u;

#pragma unroll
  for (int kw = 0; kw < CC_KW; ++kw) {
    const int g = sid - kw;
    if (g < 0) negmask |= (1u << kw);
    int gc = g < 0 ? 0 : (g > CC_LKV - 1 ? CC_LKV - 1 : g);

    const float* krow = kv + ((size_t)(b * CC_LKV + gc)) * (2 * CC_H * CC_DH) + h * CC_DH;
    const float k1 = krow[lane];
    const float k2 = krow[32 + lane];
    v1r[kw] = krow[CC_H * CC_DH + lane];
    v2r[kw] = krow[CC_H * CC_DH + 32 + lane];

    int kp = kv_pos_ids[gc];
    kp = kp < 0 ? 0 : (kp > CC_SMAX - 1 ? CC_SMAX - 1 : kp);
    const float ak = (float)kp * inv_freq;
    const float ck = __cosf(ak), sk = __sinf(ak);
    const float kr1 = k1 * ck - k2 * sk;
    const float kr2 = k2 * ck + k1 * sk;

    float part = qr1 * kr1 + qr2 * kr2;
#pragma unroll
    for (int off = 16; off > 0; off >>= 1)
      part += __shfl_xor(part, off, 32);
    sc[kw] = part * 0.125f; // DH^-0.5
  }

  float mx = -3.4e38f;
#pragma unroll
  for (int kw = 0; kw < CC_KW; ++kw)
    if (!((negmask >> kw) & 1u) && sc[kw] > mx) mx = sc[kw];

  float e[CC_KW], denom = 0.f;
#pragma unroll
  for (int kw = 0; kw < CC_KW; ++kw) {
    e[kw] = ((negmask >> kw) & 1u) ? 0.f : __expf(sc[kw] - mx);
    denom += e[kw];
  }
  const float rden = 1.0f / denom;

  float o1 = 0.f, o2 = 0.f;
#pragma unroll
  for (int kw = 0; kw < CC_KW; ++kw) {
    const float p = e[kw] * rden;
    o1 += p * v1r[kw];
    o2 += p * v2r[kw];
  }

  attn_out[qoff + lane]      = (__bf16)o1;
  attn_out[qoff + 32 + lane] = (__bf16)o2;
}

// ---------------------------------------------------------------------------
// Launch
//   d_in: 0=q(2,4096,1024) 1=kv_src(2,512,1024) 2=Wq(1024,1024)
//         3=Wkv(1024,2048) 4=Wo(1024,1024) 5=seg_id(2,4096)
//         6=q_pos_ids(2,4096) 7=kv_pos_ids(512)
//   d_out: (2,4096,1024) fp32
// ---------------------------------------------------------------------------
extern "C" void kernel_launch(void* const* d_in, const int* in_sizes, int n_in,
                              void* d_out, int out_size, void* d_ws, size_t ws_size,
                              hipStream_t stream) {
  const float* q      = (const float*)d_in[0];
  const float* kv_src = (const float*)d_in[1];
  const float* Wq     = (const float*)d_in[2];
  const float* Wkv    = (const float*)d_in[3];
  const float* Wo     = (const float*)d_in[4];
  const int*   seg_id = (const int*)d_in[5];
  const int*   q_pos  = (const int*)d_in[6];
  const int*   kv_pos = (const int*)d_in[7];
  float* out = (float*)d_out;

  const int MQ = CC_B * CC_LQ;   // 8192
  const int MK = CC_B * CC_LKV;  // 1024
  const int D  = 1024;           // model dim

  // ---- workspace carve (~86 MB) ----
  char* wsp = (char*)d_ws;
  float*  qh_ws    = (float*)wsp;  wsp += (size_t)MQ * D * 4;      // 33.6 MB
  float*  kv_ws    = (float*)wsp;  wsp += (size_t)MK * 2 * D * 4;  //  8.4 MB
  __bf16* q_bf     = (__bf16*)wsp; wsp += (size_t)MQ * D * 2;      // 16.8 MB
  __bf16* kvsrc_bf = (__bf16*)wsp; wsp += (size_t)MK * D * 2;      //  2.1 MB
  __bf16* attn_bf  = (__bf16*)wsp; wsp += (size_t)MQ * D * 2;      // 16.8 MB
  __bf16* Wqt      = (__bf16*)wsp; wsp += (size_t)D * D * 2;       //  2.1 MB
  __bf16* Wkvt     = (__bf16*)wsp; wsp += (size_t)2 * D * D * 2;   //  4.2 MB
  __bf16* Wot      = (__bf16*)wsp; wsp += (size_t)D * D * 2;       //  2.1 MB
  (void)in_sizes; (void)n_in; (void)out_size; (void)ws_size;

  // ---- prep: activations to bf16, weights to bf16 transposed [n][k] ----
  convert_f32_to_bf16<<<(MQ * D / 4) / 256, 256, 0, stream>>>(q, q_bf, MQ * D / 4);
  convert_f32_to_bf16<<<(MK * D / 4) / 256, 256, 0, stream>>>(kv_src, kvsrc_bf, MK * D / 4);
  transpose_f32_to_bf16<<<dim3(D / 32, D / 32), 256, 0, stream>>>(Wq, Wqt, D, D);
  transpose_f32_to_bf16<<<dim3(2 * D / 32, D / 32), 256, 0, stream>>>(Wkv, Wkvt, D, 2 * D);
  transpose_f32_to_bf16<<<dim3(D / 32, D / 32), 256, 0, stream>>>(Wo, Wot, D, D);

  // ---- KV projection: (1024,1024)@(1024,2048) -> kv_ws f32 ----
  gemm_bf16_wmma_async<<<dim3(2 * D / 128, MK / 128), 256, 0, stream>>>(
      kvsrc_bf, Wkvt, kv_ws, MK, 2 * D, D);

  // ---- Q projection: (8192,1024)@(1024,1024) -> qh_ws f32 ----
  gemm_bf16_wmma_async<<<dim3(D / 128, MQ / 128), 256, 0, stream>>>(
      q_bf, Wqt, qh_ws, MQ, D, D);

  // ---- attention: one wave per (b,q,h), bf16 output ----
  {
    const int waves = CC_B * CC_LQ * CC_H;   // 131072
    attn_window_kernel<<<(waves * 32) / 256, 256, 0, stream>>>(
        qh_ws, attn_bf, kv_ws, seg_id, q_pos, kv_pos);
  }

  // ---- output projection: (8192,1024)@(1024,1024) -> d_out f32 ----
  gemm_bf16_wmma_async<<<dim3(D / 128, MQ / 128), 256, 0, stream>>>(
      attn_bf, Wot, out, MQ, D, D);
}
